// RansPool_62165356642571
// MI455X (gfx1250) — matrix operand order repaired and checked by reading, not compiled, not run on hardware
//
#include <hip/hip_runtime.h>
#include <math.h>

#define HIDDEN 192
#define NNODES 262144
#define NEDGES 262144
#define NSEG   16384

typedef __attribute__((ext_vector_type(16))) _Float16 v16h;
typedef __attribute__((ext_vector_type(8)))  float    v8f;

union Frag {
    v16h  h;
    uint4 u[2];
};

// Build a 16-half fragment from two 16-byte chunks (A-fragment gather, or
// two halves of a packed 32-byte B-fragment).
__device__ __forceinline__ v16h load_frag2(const _Float16* p0, const _Float16* p1) {
    Frag f;
    f.u[0] = *reinterpret_cast<const uint4*>(p0);
    f.u[1] = *reinterpret_cast<const uint4*>(p1);
    return f.h;
}

__device__ __forceinline__ float gelu_exact(float x) {
    return 0.5f * x * (1.0f + erff(x * 0.70710678118654752f));
}

// ---------------- sincos embedding table (f32 pos -> f16 table) ----------------
__global__ void embed_kernel(const float* __restrict__ pos, _Float16* __restrict__ emb) {
    int n = blockIdx.x * blockDim.x + threadIdx.x;
    if (n >= NNODES) return;
    const float LN1E4 = 9.210340371976184f;  // ln(10000)
    _Float16* row = emb + (size_t)n * HIDDEN;
#pragma unroll
    for (int d = 0; d < 3; ++d) {
        float c = pos[n * 3 + d];
        for (int j = 0; j < 32; ++j) {
            float a   = (float)(2 * j) * (1.0f / 64.0f);
            float om  = __expf(-a * LN1E4);
            float ang = c * om;
            float s, co;
            __sincosf(ang, &s, &co);
            row[d * 64 + j]      = (_Float16)s;
            row[d * 64 + 32 + j] = (_Float16)co;
        }
    }
}

// ------------- repack f32 row-major weights into f16 B-fragment tiles ----------
// Packed layout: tile (ct,ks) occupies 512 halves; lane l holds 16 contiguous
// halves = W[ks*32 + (l<16?0:8) + (i&7) + (i>=8?16:0)][ct*16 + (l&15)], i = 0..15.
__global__ void pack_kernel(const float* __restrict__ W, _Float16* __restrict__ P,
                            int Kdim, int Ndim) {
    int t = blockIdx.x * blockDim.x + threadIdx.x;
    if (t >= Kdim * Ndim) return;
    int i      = t & 15;
    int lane   = (t >> 4) & 31;
    int tile   = t >> 9;
    int ktiles = Kdim >> 5;
    int ct     = tile / ktiles;
    int ks     = tile - ct * ktiles;
    int col    = ct * 16 + (lane & 15);
    int krow   = ks * 32 + ((lane < 16) ? 0 : 8) + (i & 7) + ((i >> 3) << 4);
    P[t] = (_Float16)W[krow * Ndim + col];
}

// -------------------- fused edge-MLP + segment-mean pool ----------------------
// 128 threads = 4 waves; each wave owns one 16-edge segment (one dst node).
__launch_bounds__(128)
__global__ void mlp_pool_kernel(const _Float16* __restrict__ emb,
                                const _Float16* __restrict__ pw1,
                                const _Float16* __restrict__ pw2,
                                const _Float16* __restrict__ pw3,
                                const int*      __restrict__ edges,
                                const float*    __restrict__ b1,
                                const float*    __restrict__ b2,
                                const float*    __restrict__ b3,
                                float*          __restrict__ out) {
    __shared__ _Float16 lds[4][16 * 392];  // per-wave activation buffer (50 KB total)

    const int lane    = threadIdx.x & 31;
    const int wave    = threadIdx.x >> 5;
    const int rl      = lane & 15;
    const int hi      = (lane >= 16) ? 1 : 0;
    const int rowBase = blockIdx.x * 64 + wave * 16;
    const int e       = rowBase + rl;          // this lane's edge row

    const int dsti = edges[2 * e];
    const int srci = edges[2 * e + 1];
    const _Float16* rowS = emb + (size_t)srci * HIDDEN;
    const _Float16* rowD = emb + (size_t)dsti * HIDDEN;
    _Float16* hb = &lds[wave][0];

    v16h afrag[12];

    // ---- gather layer-1 A fragments straight from the f16 embedding table ----
    // 16-bit A 16x32 layout: lanes<16 take K {k0..k0+7, k0+16..k0+23},
    // lanes>=16 take K {k0+8.., k0+24..}. Feature f<192 -> x[src], else x[dst].
#pragma unroll
    for (int ks = 0; ks < 12; ++ks) {
        int f0 = ks * 32 + (hi ? 8 : 0);
        int f1 = f0 + 16;
        const _Float16* p0 = (f0 < HIDDEN) ? (rowS + f0) : (rowD + (f0 - HIDDEN));
        const _Float16* p1 = (f1 < HIDDEN) ? (rowS + f1) : (rowD + (f1 - HIDDEN));
        afrag[ks] = load_frag2(p0, p1);
    }

    // ---- layer 1: (16x384)@(384x384) + b1, gelu -> LDS (row stride 392) ----
#pragma unroll 1
    for (int ct = 0; ct < 24; ++ct) {
        v8f acc = {};
        const _Float16* wt = pw1 + ((size_t)ct * 12) * 512 + lane * 16;
#pragma unroll
        for (int ks = 0; ks < 12; ++ks) {
            v16h b = load_frag2(wt, wt + 8);
            wt += 512;
            acc = __builtin_amdgcn_wmma_f32_16x16x32_f16(false, afrag[ks], false, b,
                                                         (short)0, acc, false, false);
        }
        float bias = b1[ct * 16 + rl];
        int   col  = ct * 16 + rl;
#pragma unroll
        for (int r = 0; r < 8; ++r) {
            int row = r + (hi ? 8 : 0);   // C layout: VGPR r -> M = r (+8 for hi half)
            hb[row * 392 + col] = (_Float16)gelu_exact(acc[r] + bias);
        }
    }
    __syncthreads();

    // ---- preload H1 fragments, then the LDS region can be reused for H2 ----
#pragma unroll
    for (int ks = 0; ks < 12; ++ks) {
        const _Float16* p0 = hb + rl * 392 + ks * 32 + (hi ? 8 : 0);
        afrag[ks] = load_frag2(p0, p0 + 16);
    }
    __syncthreads();

    // ---- layer 2: (16x384)@(384x192) + b2, gelu -> LDS (row stride 200) ----
#pragma unroll 1
    for (int ct = 0; ct < 12; ++ct) {
        v8f acc = {};
        const _Float16* wt = pw2 + ((size_t)ct * 12) * 512 + lane * 16;
#pragma unroll
        for (int ks = 0; ks < 12; ++ks) {
            v16h b = load_frag2(wt, wt + 8);
            wt += 512;
            acc = __builtin_amdgcn_wmma_f32_16x16x32_f16(false, afrag[ks], false, b,
                                                         (short)0, acc, false, false);
        }
        float bias = b2[ct * 16 + rl];
        int   col  = ct * 16 + rl;
#pragma unroll
        for (int r = 0; r < 8; ++r) {
            int row = r + (hi ? 8 : 0);
            hb[row * 200 + col] = (_Float16)gelu_exact(acc[r] + bias);
        }
    }
    __syncthreads();

    // ---- preload H2 fragments ----
#pragma unroll
    for (int ks = 0; ks < 6; ++ks) {
        const _Float16* p0 = hb + rl * 200 + ks * 32 + (hi ? 8 : 0);
        afrag[ks] = load_frag2(p0, p0 + 16);
    }

    // ---- layer 3: (16x192)@(192x192); fused mean over the 16-row segment ----
    const int seg = edges[2 * rowBase];  // dst shared by all 16 edges of this wave
#pragma unroll 1
    for (int ct = 0; ct < 12; ++ct) {
        v8f acc = {};
        const _Float16* wt = pw3 + ((size_t)ct * 6) * 512 + lane * 16;
#pragma unroll
        for (int ks = 0; ks < 6; ++ks) {
            v16h b = load_frag2(wt, wt + 8);
            wt += 512;
            acc = __builtin_amdgcn_wmma_f32_16x16x32_f16(false, afrag[ks], false, b,
                                                         (short)0, acc, false, false);
        }
        // sum rows 0..7 (lanes<16) / 8..15 (lanes>=16) of the C tile ...
        float s = acc[0] + acc[1] + acc[2] + acc[3] + acc[4] + acc[5] + acc[6] + acc[7];
        // ... then fold the two halves together across lanes l <-> l+16
        s += __shfl_xor(s, 16, 32);
        if (!hi) {
            out[(size_t)seg * HIDDEN + ct * 16 + lane] =
                s * (1.0f / 16.0f) + b3[ct * 16 + lane];
        }
    }
}

extern "C" void kernel_launch(void* const* d_in, const int* in_sizes, int n_in,
                              void* d_out, int out_size, void* d_ws, size_t ws_size,
                              hipStream_t stream) {
    const float* mesh_pos   = (const float*)d_in[0];
    const int*   mesh_edges = (const int*)d_in[1];
    // d_in[2] = batch_idx (unused by the math)
    const float* W1 = (const float*)d_in[3];
    const float* b1 = (const float*)d_in[4];
    const float* W2 = (const float*)d_in[5];
    const float* b2 = (const float*)d_in[6];
    const float* W3 = (const float*)d_in[7];
    const float* b3 = (const float*)d_in[8];
    float*       out = (float*)d_out;

    char* ws = (char*)d_ws;
    _Float16* emb = (_Float16*)ws;                         // 96 MB f16 table (L2-resident)
    size_t embBytes = (size_t)NNODES * HIDDEN * sizeof(_Float16);
    _Float16* pw1 = (_Float16*)(ws + embBytes);
    _Float16* pw2 = pw1 + 384 * 384;
    _Float16* pw3 = pw2 + 384 * 192;

    pack_kernel<<<(384 * 384 + 255) / 256, 256, 0, stream>>>(W1, pw1, 384, 384);
    pack_kernel<<<(384 * 192 + 255) / 256, 256, 0, stream>>>(W2, pw2, 384, 192);
    pack_kernel<<<(192 * 192 + 255) / 256, 256, 0, stream>>>(W3, pw3, 192, 192);
    embed_kernel<<<NNODES / 256, 256, 0, stream>>>(mesh_pos, emb);
    mlp_pool_kernel<<<NEDGES / 64, 128, 0, stream>>>(emb, pw1, pw2, pw3, mesh_edges,
                                                     b1, b2, b3, out);
}
